// TransformerBlock_51969104281988
// MI455X (gfx1250) — matrix operand order repaired and checked by reading, not compiled, not run on hardware
//
#include <hip/hip_runtime.h>
#include <math.h>

// ---------------- CDNA5 WMMA types ----------------
typedef __attribute__((ext_vector_type(16))) _Float16 v16h;
typedef __attribute__((ext_vector_type(8)))  float    v8f;

union Frag16 { v16h v; unsigned int u[8]; unsigned short s[16]; };
union CFrag  { v8f  v; float f[8]; };

static __device__ __forceinline__ unsigned short f2hbits(float x) {
    _Float16 h = (_Float16)x;
    unsigned short u;
    __builtin_memcpy(&u, &h, 2);
    return u;
}

// ---------------- TDM (Tensor Data Mover) support ----------------
typedef unsigned int u32x4 __attribute__((ext_vector_type(4)));
typedef int          i32x4 __attribute__((ext_vector_type(4)));
typedef int          i32x8 __attribute__((ext_vector_type(8)));

typedef __attribute__((address_space(3))) unsigned short lds_us_t;

static __device__ __forceinline__ unsigned lds_byte_off(const unsigned short* p) {
    return (unsigned)(unsigned long long)(lds_us_t*)p;   // generic -> LDS offset
}

// DMA a 2D tile (tile_rows x tile_k halfs, row stride K halfs) from global into
// LDS, with hardware padding of 4 dwords every 16 dwords so the LDS row stride
// is 40 halfs (80 B) -- identical to the hand-staged layout.
static __device__ __forceinline__ void tdm_load_tile_f16(
    const unsigned short* gtile, unsigned lds_off,
    int K /*dim0 len & stride*/, int Nrows /*dim1 len*/,
    int tile_k /*32*/, int tile_rows /*128*/)
{
    unsigned long long ga = (unsigned long long)gtile;
    u32x4 g0;
    g0.x = 1u;                                            // count=1, user mode
    g0.y = lds_off;                                       // lds_addr (bytes)
    g0.z = (unsigned)(ga & 0xFFFFFFFFu);                  // global_addr[31:0]
    g0.w = (unsigned)((ga >> 32) & 0x1FFFFFFu)            // global_addr[56:32]
         | (2u << 30);                                    // type = 2 ("image")
    i32x8 g1;
    g1[0] = (1 << 16)                                     // data_size = 2B
          | (1 << 20)                                     // pad_enable
          | (3 << 22)                                     // pad_interval: 16 dwords
          | (3 << 25);                                    // pad_amount: 4 dwords
    g1[1] = (int)(((unsigned)K & 0xFFFFu) << 16);         // tensor_dim0[15:0]
    g1[2] = (int)((((unsigned)K >> 16) & 0xFFFFu)
          | (((unsigned)Nrows & 0xFFFFu) << 16));         // dim0 hi | dim1 lo
    g1[3] = (int)((((unsigned)Nrows >> 16) & 0xFFFFu)
          | ((unsigned)tile_k << 16));                    // dim1 hi | tile_dim0
    g1[4] = tile_rows;                                    // tile_dim1 (tile_dim2=0)
    g1[5] = K;                                            // tensor_dim0_stride lo
    g1[6] = 0;
    g1[7] = 0;
    i32x4 z4 = {0, 0, 0, 0};
#if __clang_major__ >= 23
    i32x8 z8 = {0, 0, 0, 0, 0, 0, 0, 0};
    __builtin_amdgcn_tensor_load_to_lds(g0, g1, z4, z4, z8, 0);
#else
    __builtin_amdgcn_tensor_load_to_lds(g0, g1, z4, z4, 0);
#endif
}

// ---------------- problem constants ----------------
#define B_   2
#define S_   2048
#define D_   1024
#define H_   16
#define HD_  64
#define FFN_ 4096
#define T_   (B_ * S_)      // 4096 tokens

// ---------------- fp32 [KxN] -> f16 transposed [NxK] ----------------
__global__ __launch_bounds__(256) void cvtT_f32_to_f16_kernel(
    const float* __restrict__ in, unsigned short* __restrict__ out, int K, int N)
{
    __shared__ float t[32][33];
    int tx = threadIdx.x & 31, ty = threadIdx.x >> 5;
    int k0 = blockIdx.x * 32, n0 = blockIdx.y * 32;
    #pragma unroll
    for (int r = 0; r < 4; r++)
        t[ty + 8 * r][tx] = in[(size_t)(k0 + ty + 8 * r) * N + n0 + tx];
    __syncthreads();
    #pragma unroll
    for (int r = 0; r < 4; r++) {
        int n = ty + 8 * r;
        out[(size_t)(n0 + n) * K + k0 + tx] = f2hbits(t[tx][n]);
    }
}

// ---------------- LayerNorm (fp32 in) -> f16 out ----------------
__global__ __launch_bounds__(256) void ln_to_f16_kernel(
    const float* __restrict__ x, const float* __restrict__ g,
    const float* __restrict__ bta, unsigned short* __restrict__ out)
{
    int row = blockIdx.x;
    int tid = threadIdx.x;
    const float4* xr = (const float4*)(x + (size_t)row * D_);
    float4 v = xr[tid];
    __shared__ float red[256];
    red[tid] = v.x + v.y + v.z + v.w; __syncthreads();
    for (int off = 128; off > 0; off >>= 1) {
        if (tid < off) red[tid] += red[tid + off];
        __syncthreads();
    }
    float mean = red[0] * (1.0f / D_);
    __syncthreads();
    float dx = v.x - mean, dy = v.y - mean, dz = v.z - mean, dw = v.w - mean;
    red[tid] = dx*dx + dy*dy + dz*dz + dw*dw; __syncthreads();
    for (int off = 128; off > 0; off >>= 1) {
        if (tid < off) red[tid] += red[tid + off];
        __syncthreads();
    }
    float rstd = rsqrtf(red[0] * (1.0f / D_) + 1e-12f);
    float4 gv = ((const float4*)g)[tid];
    float4 bv = ((const float4*)bta)[tid];
    union { unsigned short s[4]; uint2 u2; } p;
    p.s[0] = f2hbits(gv.x * (dx * rstd) + bv.x);
    p.s[1] = f2hbits(gv.y * (dy * rstd) + bv.y);
    p.s[2] = f2hbits(gv.z * (dz * rstd) + bv.z);
    p.s[3] = f2hbits(gv.w * (dw * rstd) + bv.w);
    ((uint2*)(out + (size_t)row * D_))[tid] = p.u2;
}

// ------- tiled WMMA GEMM: C[MxN] = A[MxK] * Bt[NxK]^T -------
// TDM-staged, double-buffered LDS.
// mode 0: store f16   mode 1: exact GELU then f16   mode 2: fp32 out = resid + acc
#define BM 128
#define BN 128
#define BK 32
#define LSTR 40    // halfs; 80B row stride (TDM pad: 4 dwords per 16 dwords)

__global__ __launch_bounds__(256) void gemm_f16_kernel(
    const unsigned short* __restrict__ A, const unsigned short* __restrict__ Bt,
    const float* __restrict__ resid, void* __restrict__ outp,
    int M, int N, int K, int mode)
{
    __shared__ unsigned short As[2][BM * LSTR];
    __shared__ unsigned short Bs[2][BN * LSTR];
    int tid = threadIdx.x;
    int lane = tid & 31, wv = tid >> 5;
    int hi = lane >> 4, ln16 = lane & 15;
    int wm = wv >> 2, wn = wv & 3;          // 2 x 4 wave grid
    int bm0 = blockIdx.y * BM, bn0 = blockIdx.x * BN;

    const unsigned short* Atile = A  + (size_t)bm0 * K;   // tile origin, k advances
    const unsigned short* Btile = Bt + (size_t)bn0 * K;

    CFrag acc[4][2];
    #pragma unroll
    for (int i = 0; i < 4; i++)
        #pragma unroll
        for (int j = 0; j < 2; j++)
            #pragma unroll
            for (int r = 0; r < 8; r++) acc[i][j].f[r] = 0.0f;

    int NT = K / BK;
    // prologue: TDM tile 0 -> buffer 0 (wave 0 issues; TENSORcnt is per-wave)
    if (wv == 0) {
        tdm_load_tile_f16(Atile, lds_byte_off(&As[0][0]), K, M, BK, BM);
        tdm_load_tile_f16(Btile, lds_byte_off(&Bs[0][0]), K, N, BK, BN);
        __builtin_amdgcn_s_wait_tensorcnt(0);
    }
    __syncthreads();

    for (int it = 0; it < NT; ++it) {
        int cur = it & 1;
        bool more = (it + 1) < NT;
        if (more && wv == 0) {   // DMA next tile into the other buffer
            size_t ko = (size_t)(it + 1) * BK;
            tdm_load_tile_f16(Atile + ko, lds_byte_off(&As[1 - cur][0]), K, M, BK, BM);
            tdm_load_tile_f16(Btile + ko, lds_byte_off(&Bs[1 - cur][0]), K, N, BK, BN);
            if ((it + 2) < NT) {   // L2 prefetch two tiles ahead
                __builtin_prefetch(Atile + (size_t)(it + 2) * BK, 0, 1);
                __builtin_prefetch(Btile + (size_t)(it + 2) * BK, 0, 1);
            }
        }

        // fragments: contiguous 16B runs -> ds_load_b128
        Frag16 af[4], bf[2];
        #pragma unroll
        for (int i = 0; i < 4; i++) {
            int row = wm * 64 + i * 16 + ln16;
            const unsigned short* p0 = &As[cur][row * LSTR + hi * 16];
            #pragma unroll
            for (int p = 0; p < 4; p++) af[i].u[p] = *(const unsigned int*)(p0 + 2 * p);
            #pragma unroll
            for (int p = 0; p < 4; p++) af[i].u[4 + p] = *(const unsigned int*)(p0 + 8 + 2 * p);
        }
        #pragma unroll
        for (int j = 0; j < 2; j++) {
            int col = wn * 32 + j * 16 + ln16;
            const unsigned short* p0 = &Bs[cur][col * LSTR + hi * 16];
            #pragma unroll
            for (int p = 0; p < 8; p++) bf[j].u[p] = *(const unsigned int*)(p0 + 2 * p);
        }
        #pragma unroll
        for (int i = 0; i < 4; i++)
            #pragma unroll
            for (int j = 0; j < 2; j++)
                acc[i][j].v = __builtin_amdgcn_wmma_f32_16x16x32_f16(
                    false, af[i].v, false, bf[j].v, (short)0, acc[i][j].v, false, false);

        if (more) {
            if (wv == 0) __builtin_amdgcn_s_wait_tensorcnt(0);
            __syncthreads();   // next buffer complete & visible; old reads done
        }
    }

    // epilogue: C layout row = r + 8*hi, col = lane&15
    #pragma unroll
    for (int i = 0; i < 4; i++)
        #pragma unroll
        for (int j = 0; j < 2; j++) {
            int row0 = bm0 + wm * 64 + i * 16 + 8 * hi;
            int col  = bn0 + wn * 32 + j * 16 + ln16;
            if (mode == 2) {
                float rv[8];
                #pragma unroll
                for (int r = 0; r < 8; r++)
                    rv[r] = resid[(size_t)(row0 + r) * N + col];
                #pragma unroll
                for (int r = 0; r < 8; r++)
                    ((float*)outp)[(size_t)(row0 + r) * N + col] = rv[r] + acc[i][j].f[r];
            } else {
                #pragma unroll
                for (int r = 0; r < 8; r++) {
                    float vv = acc[i][j].f[r];
                    if (mode == 1)
                        vv = 0.5f * vv * (1.0f + erff(vv * 0.70710678118654752f));
                    ((unsigned short*)outp)[(size_t)(row0 + r) * N + col] = f2hbits(vv);
                }
            }
        }
}

// ---------------- flash-style attention ----------------
// grid: (S/64, B*H), block: 128 threads (4 waves, 16 query rows each)
#define KSTR 72
#define PSTR 72

__global__ __launch_bounds__(128) void attn_kernel(
    const unsigned short* __restrict__ qg, const unsigned short* __restrict__ kg,
    const unsigned short* __restrict__ vg, const float* __restrict__ mask,
    unsigned short* __restrict__ outg)
{
    __shared__ unsigned short Ks[64 * KSTR];          // K tile [key][dim]
    __shared__ unsigned short Vt[64 * KSTR];          // V tile transposed [dim][key]
    __shared__ unsigned short Ps[4 * 16 * PSTR];      // per-wave P tiles (f16)

    int tid = threadIdx.x, lane = tid & 31, wv = tid >> 5;
    int hi = lane >> 4, ln16 = lane & 15;
    int qblk = blockIdx.x, bh = blockIdx.y;
    int b = bh >> 4, h = bh & 15;
    size_t tb = (size_t)b * S_;
    int q0 = qblk * 64 + wv * 16;

    // Q A-fragments (K = head_dim 64 -> two 16x32 frags), kept in registers
    Frag16 aq[2];
    {
        size_t base = (tb + q0 + ln16) * D_ + h * HD_;
        #pragma unroll
        for (int j = 0; j < 2; j++)
            #pragma unroll
            for (int p = 0; p < 8; p++) {
                int kk = hi * 16 + 2 * p + j * 32;
                aq[j].u[p] = *(const unsigned int*)(qg + base + kk);
            }
    }

    float mrow[8], lrow[8];
    CFrag o[4];
    #pragma unroll
    for (int r = 0; r < 8; r++) { mrow[r] = -3.0e38f; lrow[r] = 0.0f; }
    #pragma unroll
    for (int t = 0; t < 4; t++)
        #pragma unroll
        for (int r = 0; r < 8; r++) o[t].f[r] = 0.0f;

    for (int kb = 0; kb < S_ / 64; kb++) {
        __syncthreads();
        // stage 64x64 K tile and transposed V tile
        #pragma unroll
        for (int c = tid; c < 512; c += 128) {
            int key = c >> 3, cc = (c & 7) << 3;
            size_t ga = (tb + (size_t)kb * 64 + key) * D_ + h * HD_ + cc;
            float4 tk = *(const float4*)(kg + ga);
            *(float4*)(&Ks[key * KSTR + cc]) = tk;
            float4 tv = *(const float4*)(vg + ga);
            union { float4 f4; unsigned short s[8]; } uv; uv.f4 = tv;
            #pragma unroll
            for (int jj = 0; jj < 8; jj++) Vt[(cc + jj) * KSTR + key] = uv.s[jj];
        }
        if (kb + 1 < S_ / 64) {
            size_t gn = (tb + (size_t)(kb + 1) * 64 + (tid >> 1)) * D_ + h * HD_;
            __builtin_prefetch(kg + gn, 0, 1);
            __builtin_prefetch(vg + gn, 0, 1);
        }
        __syncthreads();

        // scores: 4 key sub-tiles of 16, each = 2 WMMAs over head_dim
        CFrag sc[4];
        #pragma unroll
        for (int t = 0; t < 4; t++) {
            v8f a = {0.f,0.f,0.f,0.f,0.f,0.f,0.f,0.f};
            int keyc = t * 16 + ln16;
            #pragma unroll
            for (int ks = 0; ks < 2; ks++) {
                Frag16 bk;
                const unsigned short* p0 = &Ks[keyc * KSTR + hi * 16 + ks * 32];
                #pragma unroll
                for (int p = 0; p < 8; p++) bk.u[p] = *(const unsigned int*)(p0 + 2 * p);
                a = __builtin_amdgcn_wmma_f32_16x16x32_f16(
                        false, aq[ks].v, false, bk.v, (short)0, a, false, false);
            }
            sc[t].v = a;
            int kcol = kb * 64 + t * 16 + ln16;
            #pragma unroll
            for (int r = 0; r < 8; r++) {
                int qrow = q0 + r + 8 * hi;
                sc[t].f[r] = sc[t].f[r] * 0.125f + mask[(size_t)qrow * S_ + kcol];
            }
        }

        // online softmax (per-row scalars replicated across 16-lane group)
        #pragma unroll
        for (int r = 0; r < 8; r++) {
            float vmax = sc[0].f[r];
            #pragma unroll
            for (int t = 1; t < 4; t++) vmax = fmaxf(vmax, sc[t].f[r]);
            for (int mm = 1; mm < 16; mm <<= 1) vmax = fmaxf(vmax, __shfl_xor(vmax, mm, 32));
            float mnew = fmaxf(mrow[r], vmax);
            float fsc = expf(mrow[r] - mnew);
            mrow[r] = mnew;
            lrow[r] *= fsc;
            #pragma unroll
            for (int t = 0; t < 4; t++) o[t].f[r] *= fsc;
            float s = 0.0f;
            #pragma unroll
            for (int t = 0; t < 4; t++) {
                float p = expf(sc[t].f[r] - mnew);
                sc[t].f[r] = p;
                s += p;
            }
            for (int mm = 1; mm < 16; mm <<= 1) s += __shfl_xor(s, mm, 32);
            lrow[r] += s;
        }

        // re-quantize P to f16 through LDS (wave-private region)
        #pragma unroll
        for (int t = 0; t < 4; t++)
            #pragma unroll
            for (int r = 0; r < 8; r++)
                Ps[(wv * 16 + r + 8 * hi) * PSTR + t * 16 + ln16] = f2hbits(sc[t].f[r]);

        // O += P @ V : 2 k-steps x 4 dim tiles
        #pragma unroll
        for (int ks = 0; ks < 2; ks++) {
            Frag16 ap;
            const unsigned short* pa = &Ps[(wv * 16 + ln16) * PSTR + ks * 32 + hi * 16];
            #pragma unroll
            for (int p = 0; p < 4; p++) ap.u[p] = *(const unsigned int*)(pa + 2 * p);
            #pragma unroll
            for (int p = 0; p < 4; p++) ap.u[4 + p] = *(const unsigned int*)(pa + 8 + 2 * p);
            #pragma unroll
            for (int t = 0; t < 4; t++) {
                Frag16 bv;
                const unsigned short* pb = &Vt[(t * 16 + ln16) * KSTR + hi * 16 + ks * 32];
                #pragma unroll
                for (int p = 0; p < 8; p++) bv.u[p] = *(const unsigned int*)(pb + 2 * p);
                o[t].v = __builtin_amdgcn_wmma_f32_16x16x32_f16(
                            false, ap.v, false, bv.v, (short)0, o[t].v, false, false);
            }
        }
    }

    // normalize and write attn output (f16, [B,S,DIM] head slice)
    #pragma unroll
    for (int t = 0; t < 4; t++)
        #pragma unroll
        for (int r = 0; r < 8; r++) {
            size_t tok = tb + q0 + r + 8 * hi;
            outg[tok * D_ + h * HD_ + t * 16 + ln16] = f2hbits(o[t].f[r] / lrow[r]);
        }
}

// ---------------- host-side orchestration ----------------
extern "C" void kernel_launch(void* const* d_in, const int* in_sizes, int n_in,
                              void* d_out, int out_size, void* d_ws, size_t ws_size,
                              hipStream_t stream) {
    const float* x     = (const float*)d_in[0];
    const float* mask  = (const float*)d_in[1];
    const float* wq    = (const float*)d_in[2];
    const float* wk    = (const float*)d_in[3];
    const float* wv    = (const float*)d_in[4];
    const float* wo    = (const float*)d_in[5];
    const float* w1    = (const float*)d_in[6];
    const float* w2    = (const float*)d_in[7];
    const float* ln1w  = (const float*)d_in[8];
    const float* ln1b  = (const float*)d_in[9];
    const float* ln2w  = (const float*)d_in[10];
    const float* ln2b  = (const float*)d_in[11];
    float* out = (float*)d_out;

    char* ws = (char*)d_ws;
    const size_t MB = 1024 * 1024;
    unsigned short* wq_h   = (unsigned short*)(ws + 0 * MB);   // transposed [N x K] f16
    unsigned short* wk_h   = (unsigned short*)(ws + 2 * MB);
    unsigned short* wv_h   = (unsigned short*)(ws + 4 * MB);
    unsigned short* wo_h   = (unsigned short*)(ws + 6 * MB);
    unsigned short* w1_h   = (unsigned short*)(ws + 8 * MB);
    unsigned short* w2_h   = (unsigned short*)(ws + 16 * MB);
    unsigned short* h_h    = (unsigned short*)(ws + 24 * MB);
    unsigned short* q_h    = (unsigned short*)(ws + 32 * MB);
    unsigned short* k_h    = (unsigned short*)(ws + 40 * MB);
    unsigned short* v_h    = (unsigned short*)(ws + 48 * MB);
    unsigned short* attn_h = (unsigned short*)(ws + 56 * MB);
    float*          xattn  = (float*)        (ws + 64 * MB);
    unsigned short* h2_h   = (unsigned short*)(ws + 80 * MB);
    unsigned short* act_h  = (unsigned short*)(ws + 88 * MB);

    // 1) weights fp32 [KxN] -> transposed f16 [NxK]
    cvtT_f32_to_f16_kernel<<<dim3(D_/32,  D_/32 ), 256, 0, stream>>>(wq, wq_h, D_,  D_);
    cvtT_f32_to_f16_kernel<<<dim3(D_/32,  D_/32 ), 256, 0, stream>>>(wk, wk_h, D_,  D_);
    cvtT_f32_to_f16_kernel<<<dim3(D_/32,  D_/32 ), 256, 0, stream>>>(wv, wv_h, D_,  D_);
    cvtT_f32_to_f16_kernel<<<dim3(D_/32,  D_/32 ), 256, 0, stream>>>(wo, wo_h, D_,  D_);
    cvtT_f32_to_f16_kernel<<<dim3(D_/32,  FFN_/32), 256, 0, stream>>>(w1, w1_h, D_,  FFN_);
    cvtT_f32_to_f16_kernel<<<dim3(FFN_/32, D_/32 ), 256, 0, stream>>>(w2, w2_h, FFN_, D_);

    // 2) LN1 -> f16 activations
    ln_to_f16_kernel<<<T_, 256, 0, stream>>>(x, ln1w, ln1b, h_h);

    // 3) QKV projections
    dim3 gq(D_ / BN, T_ / BM);   // (8, 32)
    gemm_f16_kernel<<<gq, 256, 0, stream>>>(h_h, wq_h, nullptr, q_h, T_, D_, D_, 0);
    gemm_f16_kernel<<<gq, 256, 0, stream>>>(h_h, wk_h, nullptr, k_h, T_, D_, D_, 0);
    gemm_f16_kernel<<<gq, 256, 0, stream>>>(h_h, wv_h, nullptr, v_h, T_, D_, D_, 0);

    // 4) attention
    attn_kernel<<<dim3(S_ / 64, B_ * H_), 128, 0, stream>>>(q_h, k_h, v_h, mask, attn_h);

    // 5) output projection + residual -> fp32
    gemm_f16_kernel<<<gq, 256, 0, stream>>>(attn_h, wo_h, x, xattn, T_, D_, D_, 2);

    // 6) LN2 -> f16
    ln_to_f16_kernel<<<T_, 256, 0, stream>>>(xattn, ln2w, ln2b, h2_h);

    // 7) FFN1 + exact GELU
    dim3 g1(FFN_ / BN, T_ / BM); // (32, 32)
    gemm_f16_kernel<<<g1, 256, 0, stream>>>(h2_h, w1_h, nullptr, act_h, T_, FFN_, D_, 1);

    // 8) FFN2 + residual -> d_out (fp32)
    gemm_f16_kernel<<<gq, 256, 0, stream>>>(act_h, w2_h, xattn, out, T_, D_, FFN_, 2);

    (void)in_sizes; (void)n_in; (void)out_size; (void)ws_size;
}
